// SA_Module_214748365416
// MI455X (gfx1250) — matrix operand order repaired and checked by reading, not compiled, not run on hardware
//
#include <hip/hip_runtime.h>

typedef _Float16 f16;
typedef __attribute__((ext_vector_type(16))) _Float16 v16h;
typedef __attribute__((ext_vector_type(8)))  _Float16 v8h;
typedef __attribute__((ext_vector_type(8)))  float    v8f;

#define B_    8
#define N_    8192
#define C_    64
#define M_    2048
#define K_    32
#define DIN   67
#define DINP  96
#define H_    128
#define OUT_  256

// ---------------------------------------------------------------------------
// Weight conversion: f32 row-major [k][n]  ->  f16 transposed [n][k] (k padded)
// ws layout: W1t[H_][DINP] | W2t[OUT_][H_] | Wst[OUT_][DINP]
// ---------------------------------------------------------------------------
__global__ void convert_weights(const float* __restrict__ W1,
                                const float* __restrict__ W2,
                                const float* __restrict__ Ws,
                                f16* __restrict__ wt) {
  int i = blockIdx.x * blockDim.x + threadIdx.x;
  const int n1 = H_ * DINP;
  const int n2 = OUT_ * H_;
  const int n3 = OUT_ * DINP;
  if (i < n1) {
    int n = i / DINP, k = i % DINP;
    wt[i] = (k < DIN) ? (f16)W1[k * H_ + n] : (f16)0.f;
  } else if (i < n1 + n2) {
    int j = i - n1;
    int n = j / H_, k = j % H_;
    wt[i] = (f16)W2[k * OUT_ + n];
  } else if (i < n1 + n2 + n3) {
    int j = i - n1 - n2;
    int n = j / DINP, k = j % DINP;
    wt[i] = (k < DIN) ? (f16)Ws[k * OUT_ + n] : (f16)0.f;
  }
}

// ---------------------------------------------------------------------------
// Farthest point sampling: one block per cloud, pos + dmin resident in LDS.
// ---------------------------------------------------------------------------
__global__ __launch_bounds__(1024) void fps_kernel(const float* __restrict__ pos,
                                                   int* __restrict__ idx_out,
                                                   float* __restrict__ pos_out,
                                                   long long* __restrict__ batch_out) {
  extern __shared__ char smem[];
  float* px   = (float*)smem;        // N
  float* py   = px + N_;
  float* pz   = py + N_;
  float* dmin = pz + N_;             // N
  float* rd   = dmin + N_;           // 32
  int*   ri   = (int*)(rd + 32);     // 32
  int*   idxl = ri + 32;             // M
  int*   bcast = idxl + M_;          // 1

  const int b = blockIdx.x;
  const int t = threadIdx.x;
  const float* pb = pos + (size_t)b * N_ * 3;
  for (int j = t; j < N_; j += 1024) {
    px[j] = pb[j * 3 + 0];
    py[j] = pb[j * 3 + 1];
    pz[j] = pb[j * 3 + 2];
    dmin[j] = 1e10f;
  }
  __syncthreads();

  int last = 0;
  for (int it = 0; it < M_; ++it) {
    if (t == 0) idxl[it] = last;
    const float lx = px[last], ly = py[last], lz = pz[last];
    float bd = -1.f;
    int   bj = 0;
#pragma unroll
    for (int k = 0; k < N_ / 1024; ++k) {
      int j = t + k * 1024;
      float dx = px[j] - lx, dy = py[j] - ly, dz = pz[j] - lz;
      float d = dx * dx + dy * dy + dz * dz;
      float dm = fminf(dmin[j], d);
      dmin[j] = dm;
      if (dm > bd) { bd = dm; bj = j; }
    }
#pragma unroll
    for (int off = 16; off >= 1; off >>= 1) {
      float od = __shfl_xor(bd, off, 32);
      int   oj = __shfl_xor(bj, off, 32);
      if (od > bd || (od == bd && oj < bj)) { bd = od; bj = oj; }
    }
    int w = t >> 5;
    if ((t & 31) == 0) { rd[w] = bd; ri[w] = bj; }
    __syncthreads();
    if (w == 0) {
      bd = rd[t & 31];
      bj = ri[t & 31];
#pragma unroll
      for (int off = 16; off >= 1; off >>= 1) {
        float od = __shfl_xor(bd, off, 32);
        int   oj = __shfl_xor(bj, off, 32);
        if (od > bd || (od == bd && oj < bj)) { bd = od; bj = oj; }
      }
      if (t == 0) bcast[0] = bj;
    }
    __syncthreads();
    last = bcast[0];
  }

  for (int i = t; i < M_; i += 1024) {
    int ci = idxl[i];
    int gm = b * M_ + i;
    idx_out[gm] = ci;
    pos_out[gm * 3 + 0] = px[ci];
    pos_out[gm * 3 + 1] = py[ci];
    pos_out[gm * 3 + 2] = pz[ci];
    batch_out[gm] = (long long)b;
  }
}

// ---------------------------------------------------------------------------
// kNN: one thread per center, cloud in LDS, top-32 replace-max in LDS
// (slot-major layout -> bank-conflict-free).
// ---------------------------------------------------------------------------
__global__ __launch_bounds__(256) void knn_kernel(const float* __restrict__ pos,
                                                  const int* __restrict__ idx_ws,
                                                  int* __restrict__ nn_ws) {
  extern __shared__ char smem[];
  float* px = (float*)smem;
  float* py = px + N_;
  float* pz = py + N_;
  float* td = pz + N_;                 // [K_][256]
  int*   ti = (int*)(td + K_ * 256);   // [K_][256]

  const int b = blockIdx.x >> 3;
  const int mchunk = (blockIdx.x & 7) * 256;
  const int t = threadIdx.x;
  const float* pb = pos + (size_t)b * N_ * 3;
  for (int j = t; j < N_; j += 256) {
    px[j] = pb[j * 3 + 0];
    py[j] = pb[j * 3 + 1];
    pz[j] = pb[j * 3 + 2];
  }
#pragma unroll
  for (int s = 0; s < K_; ++s) td[s * 256 + t] = 3.4e38f;
  __syncthreads();

  const int m = mchunk + t;
  const int ci = idx_ws[b * M_ + m];
  const float cx = px[ci], cy = py[ci], cz = pz[ci];
  float curmax = 3.4e38f;
  int maxslot = 0;
  for (int j = 0; j < N_; ++j) {
    float dx = px[j] - cx, dy = py[j] - cy, dz = pz[j] - cz;
    float d = dx * dx + dy * dy + dz * dz;
    if (d < curmax) {
      td[maxslot * 256 + t] = d;
      ti[maxslot * 256 + t] = j;
      float nm = -1.f; int ns = 0;
#pragma unroll
      for (int s = 0; s < K_; ++s) {
        float v = td[s * 256 + t];
        if (v > nm) { nm = v; ns = s; }
      }
      curmax = nm; maxslot = ns;
    }
  }
  for (int s = 0; s < K_; ++s)
    nn_ws[((size_t)(b * M_ + m)) * K_ + s] = ti[s * 256 + t];
}

// ---------------------------------------------------------------------------
// WMMA fragment loaders (16-bit A 16x32 / B 32x16, contiguous k = 16*hs + s in both A and B)
// ---------------------------------------------------------------------------
__device__ __forceinline__ v16h load_a(const f16* base, int ld, int m0, int kc,
                                       int lr, int hs) {
  const f16* p = base + (m0 + lr) * ld + kc * 32 + hs * 16;
  return *(const v16h*)p;
}

__device__ __forceinline__ v16h load_b(const f16* wtT, int kdim, int tn, int kc,
                                       int lr, int hs) {
  return *(const v16h*)(wtT + (tn * 16 + lr) * kdim + kc * 32 + hs * 16);
}

// ---------------------------------------------------------------------------
// Fused gather + ResMLP + max-pool. 256 thr = 8 waves; block = 4 centers
// (128 rows). Stage1: 16x128 per wave (24 WMMA). Stage2: 16x256 per wave
// accumulating h@W2 + feat@Ws (112 WMMA).
// ---------------------------------------------------------------------------
__global__ __launch_bounds__(256) void sa_mlp_kernel(
    const float* __restrict__ x, const float* __restrict__ pos,
    const int* __restrict__ idx_ws, const int* __restrict__ nn_ws,
    const f16* __restrict__ wt, const float* __restrict__ b1,
    const float* __restrict__ b2, const float* __restrict__ bs,
    float* __restrict__ x_out) {
  extern __shared__ char smem[];
  f16* W1t   = (f16*)smem;                 // [H_][DINP]
  f16* W2t   = W1t + H_ * DINP;            // [OUT_][H_]
  f16* Wst   = W2t + OUT_ * H_;            // [OUT_][DINP]
  f16* featL = Wst + OUT_ * DINP;          // [128][DINP]
  f16* hL    = featL + 128 * DINP;         // [128][H_]
  float* cmax = (float*)(hL + 128 * H_);   // [4][2][OUT_]

  const int t = threadIdx.x;

  { // copy pre-transposed f16 weights into LDS, 16B vectors
    const int4* src = (const int4*)wt;
    int4* dst = (int4*)W1t;
    const int total = (H_ * DINP + OUT_ * H_ + OUT_ * DINP) / 8;
    for (int i = t; i < total; i += 256) dst[i] = src[i];
  }

  { // gather feat rows: 2 threads per row
    const int r = t >> 1;
    const int half = t & 1;
    const int gm = blockIdx.x * 4 + (r >> 5);
    const int s = r & 31;
    const int b = gm / M_;
    const int nj = nn_ws[(size_t)gm * K_ + s];
    const int ci = idx_ws[gm];
    const float* xr = x + ((size_t)b * N_ + nj) * C_;
    f16* frow = featL + r * DINP;
    if (half == 0) {
      for (int c = 0; c < 48; ++c) frow[c] = (f16)xr[c];
    } else {
      for (int c = 48; c < C_; ++c) frow[c] = (f16)xr[c];
      const float* pj = pos + ((size_t)b * N_ + nj) * 3;
      const float* pc = pos + ((size_t)b * N_ + ci) * 3;
      frow[64] = (f16)(pj[0] - pc[0]);
      frow[65] = (f16)(pj[1] - pc[1]);
      frow[66] = (f16)(pj[2] - pc[2]);
      for (int c = DIN; c < DINP; ++c) frow[c] = (f16)0.f;
    }
  }
  __syncthreads();

  const int w  = t >> 5;
  const int l  = t & 31;
  const int m0 = w * 16;
  const int lr = l & 15;
  const int hs = l >> 4;

  // ---- Stage 1: h = relu(feat @ W1 + b1) ----
  v8f acc1[8] = {};
#pragma unroll
  for (int kc = 0; kc < 3; ++kc) {
    v16h a = load_a(featL, DINP, m0, kc, lr, hs);
#pragma unroll
    for (int tn = 0; tn < 8; ++tn) {
      v16h bf = load_b(W1t, DINP, tn, kc, lr, hs);
      acc1[tn] = __builtin_amdgcn_wmma_f32_16x16x32_f16(
          false, a, false, bf, (short)0, acc1[tn], false, false);
    }
  }
#pragma unroll
  for (int tn = 0; tn < 8; ++tn) {
    const int col = tn * 16 + lr;
    const float bb = b1[col];
#pragma unroll
    for (int e = 0; e < 8; ++e) {
      float v = fmaxf(acc1[tn][e] + bb, 0.f);
      hL[(m0 + hs * 8 + e) * H_ + col] = (f16)v;
    }
  }
  __syncthreads();

  // ---- Stage 2: msg = relu(h @ W2 + feat @ Ws + b2 + bs) ----
  v8f acc2[16] = {};
#pragma unroll
  for (int kc = 0; kc < 4; ++kc) {
    v16h a = load_a(hL, H_, m0, kc, lr, hs);
#pragma unroll
    for (int tn = 0; tn < 16; ++tn) {
      v16h bf = load_b(W2t, H_, tn, kc, lr, hs);
      acc2[tn] = __builtin_amdgcn_wmma_f32_16x16x32_f16(
          false, a, false, bf, (short)0, acc2[tn], false, false);
    }
  }
#pragma unroll
  for (int kc = 0; kc < 3; ++kc) {
    v16h a = load_a(featL, DINP, m0, kc, lr, hs);
#pragma unroll
    for (int tn = 0; tn < 16; ++tn) {
      v16h bf = load_b(Wst, DINP, tn, kc, lr, hs);
      acc2[tn] = __builtin_amdgcn_wmma_f32_16x16x32_f16(
          false, a, false, bf, (short)0, acc2[tn], false, false);
    }
  }

  // ---- Epilogue: bias, relu, max over K=32 rows per center ----
  const int center = w >> 1;
  const int halfw  = w & 1;
#pragma unroll
  for (int tn = 0; tn < 16; ++tn) {
    const int col = tn * 16 + lr;
    const float bb = b2[col] + bs[col];
    float rmax = -3.4e38f;
#pragma unroll
    for (int e = 0; e < 8; ++e) {
      float v = fmaxf(acc2[tn][e] + bb, 0.f);
      rmax = fmaxf(rmax, v);
    }
    rmax = fmaxf(rmax, __shfl_xor(rmax, 16, 32));   // combine row halves
    if (l < 16) cmax[(center * 2 + halfw) * OUT_ + col] = rmax;
  }
  __syncthreads();

#pragma unroll
  for (int c = 0; c < 4; ++c) {
    float v = fmaxf(cmax[(c * 2 + 0) * OUT_ + t], cmax[(c * 2 + 1) * OUT_ + t]);
    int gm = blockIdx.x * 4 + c;
    x_out[(size_t)gm * OUT_ + t] = v;
  }
}

// ---------------------------------------------------------------------------
extern "C" void kernel_launch(void* const* d_in, const int* in_sizes, int n_in,
                              void* d_out, int out_size, void* d_ws, size_t ws_size,
                              hipStream_t stream) {
  const float* x   = (const float*)d_in[0];
  const float* pos = (const float*)d_in[1];
  // d_in[2]: batch (unused; batch_out regenerated)
  const float* W1 = (const float*)d_in[3];
  const float* b1 = (const float*)d_in[4];
  const float* W2 = (const float*)d_in[5];
  const float* b2 = (const float*)d_in[6];
  const float* Ws = (const float*)d_in[7];
  const float* bs = (const float*)d_in[8];

  float* out = (float*)d_out;
  float* x_out   = out;
  float* pos_out = out + (size_t)B_ * M_ * OUT_;
  long long* batch_out = (long long*)(pos_out + (size_t)B_ * M_ * 3);

  char* ws = (char*)d_ws;
  int* idx_ws = (int*)ws;                       // B*M
  int* nn_ws  = idx_ws + B_ * M_;               // B*M*K
  f16* wt     = (f16*)(nn_ws + (size_t)B_ * M_ * K_);

  const int wtotal = H_ * DINP + OUT_ * H_ + OUT_ * DINP;
  convert_weights<<<(wtotal + 255) / 256, 256, 0, stream>>>(W1, W2, Ws, wt);

  const size_t fps_smem = (size_t)(4 * N_) * 4 + 32 * 4 + 32 * 4 + M_ * 4 + 16;
  fps_kernel<<<B_, 1024, fps_smem, stream>>>(pos, idx_ws, pos_out, batch_out);

  const size_t knn_smem = (size_t)(3 * N_) * 4 + (size_t)K_ * 256 * 4 * 2;
  knn_kernel<<<B_ * (M_ / 256), 256, knn_smem, stream>>>(pos, idx_ws, nn_ws);

  const size_t mlp_smem =
      (size_t)(H_ * DINP + OUT_ * H_ + OUT_ * DINP + 128 * DINP + 128 * H_) * 2 +
      (size_t)4 * 2 * OUT_ * 4;
  sa_mlp_kernel<<<(B_ * M_) / 4, 256, mlp_smem, stream>>>(
      x, pos, idx_ws, nn_ws, wt, b1, b2, bs, x_out);
}
